// Net_global_SEDv3_90486370993038
// MI455X (gfx1250) — matrix operand order repaired and checked by reading, not compiled, not run on hardware
//
#include <hip/hip_runtime.h>
#include <cstdint>

typedef __attribute__((ext_vector_type(16))) __bf16 v16bf;
typedef __attribute__((ext_vector_type(8)))  float  v8f;

union Frag {
  v16bf  v;
  uint4  q[2];
  __bf16 e[16];
};

__device__ __forceinline__ v8f wmma_bf16(Frag a, Frag b, v8f c) {
  // (neg_a, A, neg_b, B, c_mod, C, reuse_a, reuse_b)
  return __builtin_amdgcn_wmma_f32_16x16x32_bf16(false, a.v, false, b.v, (short)0, c,
                                                 false, false);
}

// A fragment (16x32 bf16): lane holds row m=lane&15; elements 0..7 at k0+base0,
// elements 8..15 at k0+base0+16 (base0 = lane>=16 ? 8 : 0).
__device__ __forceinline__ Frag load_a(const __bf16* p) {
  Frag f;
  f.q[0] = *(const uint4*)(p);
  f.q[1] = *(const uint4*)(p + 16);
  return f;
}
// B fragment (32x16 bf16): lane holds col n=lane&15; elements 0..7 at k0+bbase,
// elements 8..15 at k0+bbase+16 (bbase = lane>=16 ? 8 : 0).
__device__ __forceinline__ Frag load_b(const __bf16* p) {
  Frag f;
  f.q[0] = *(const uint4*)(p);
  f.q[1] = *(const uint4*)(p + 16);
  return f;
}

// Per-iteration laundered zero: makes selected addresses loop-variant so LICM
// cannot hoist those loads (prevents VGPR over-subscription), while keeping
// the pointer's global address space (global_load, not flat_load).
__device__ __forceinline__ int opaque_zero() {
  int z = 0;
  asm volatile("" : "+v"(z));
  return z;
}

// CDNA5 async global->LDS copy (ASYNCcnt-tracked), 16B per lane.
__device__ __forceinline__ void async_g2l_b128(uint32_t lds_off, const void* gptr) {
  unsigned long long ga = (unsigned long long)gptr;
  asm volatile("global_load_async_to_lds_b128 %0, %1, off" : : "v"(lds_off), "v"(ga));
}
__device__ __forceinline__ void wait_asynccnt0() {
  asm volatile("s_wait_asynccnt 0x0");
}

__device__ __forceinline__ float sigm(float x) { return 1.0f / (1.0f + __expf(-x)); }

// ---------------------------------------------------------------------------
// Prep kernels: weight bf16 conversion + input projections
// ---------------------------------------------------------------------------
__global__ void prep_weights(const float* __restrict__ eWih, const float* __restrict__ eWhh,
                             const float* __restrict__ dWih, const float* __restrict__ dWhh,
                             __bf16* __restrict__ eWihb, __bf16* __restrict__ eWhhb,
                             __bf16* __restrict__ dWihb, __bf16* __restrict__ dWhhb) {
  int i = blockIdx.x * blockDim.x + threadIdx.x;
  if (i < 768 * 32) eWihb[i] = (__bf16)eWih[i];
  if (i < 768 * 256) {
    eWhhb[i] = (__bf16)eWhh[i];
    dWhhb[i] = (__bf16)dWhh[i];
  }
  if (i < 768 * 64) {  // dec Wih padded K 33 -> 64
    int r = i >> 6, c = i & 63;
    dWihb[i] = (c < 33) ? (__bf16)dWih[r * 33 + c] : (__bf16)0.0f;
  }
}

// enc_in[b][l][0:16]=X, [16:32]=Xg@Wg.T+bg, stored bf16
__global__ void prep_enc_in(const float* __restrict__ X, const float* __restrict__ Xg,
                            const float* __restrict__ Wg, const float* __restrict__ bg,
                            __bf16* __restrict__ out) {
  int i = blockIdx.x * blockDim.x + threadIdx.x;  // b*1024 + l
  if (i >= 256 * 1024) return;
  const float* x  = X  + (size_t)i * 16;
  const float* xg = Xg + (size_t)i * 32;
  __bf16* o = out + (size_t)i * 32;
  for (int f = 0; f < 16; ++f) o[f] = (__bf16)x[f];
  for (int r = 0; r < 16; ++r) {
    float s = bg[r];
    for (int g = 0; g < 32; ++g) s += xg[g] * Wg[r * 32 + g];
    o[16 + r] = (__bf16)s;
  }
}

// dec_wg[b][t][0:16]=X_dec, [16:32]=Xg_dec@Wg_dec.T+bg, stored f32
__global__ void prep_dec_wg(const float* __restrict__ X, const float* __restrict__ Xg,
                            const float* __restrict__ Wg, const float* __restrict__ bg,
                            float* __restrict__ out) {
  int i = blockIdx.x * blockDim.x + threadIdx.x;  // b*48 + t
  if (i >= 256 * 48) return;
  const float* x  = X  + (size_t)i * 16;
  const float* xg = Xg + (size_t)i * 32;
  float* o = out + (size_t)i * 32;
  for (int f = 0; f < 16; ++f) o[f] = x[f];
  for (int r = 0; r < 16; ++r) {
    float s = bg[r];
    for (int g = 0; g < 32; ++g) s += xg[g] * Wg[r * 32 + g];
    o[16 + r] = s;
  }
}

// ---------------------------------------------------------------------------
// Encoder recurrence: 16 blocks x 16 batch rows, 1024 sequential steps.
// h ping-pongs in LDS (bf16); x tile for t+1 streams in via async-to-LDS.
// Wave w owns n-tile j=w. r/z-gate Whh fragments are explicitly pre-loaded
// into a register array (128 VGPRs, loop-resident); Wih + n-gate Whh
// fragments re-fetch from L0/L2 each step via a laundered zero offset.
// ---------------------------------------------------------------------------
__global__ __launch_bounds__(512) void enc_recurrence(
    const __bf16* __restrict__ xin,   // [256][1024][32] bf16
    const __bf16* __restrict__ wihb,  // [768][32] bf16
    const __bf16* __restrict__ whhb,  // [768][256] bf16
    const float* __restrict__ bih, const float* __restrict__ bhh,
    float* __restrict__ hout)         // [256][256] f32 (final h)
{
  __shared__ __attribute__((aligned(16))) __bf16 hsh[2][16 * 256];
  __shared__ __attribute__((aligned(16))) __bf16 xsh[2][16 * 32];
  const int tid  = threadIdx.x;
  const int lane = tid & 31;
  const int w    = tid >> 5;          // 0..15 -> n-tile j within a gate
  const int mlo  = lane & 15;
  const int hi   = lane >> 4;
  const int base0 = hi ? 8 : 0;       // A-frag K sub-offset
  const int bbase = hi ? 8 : 0;       // B-frag K sub-offset
  const int m0 = blockIdx.x * 16;

  for (int i = tid; i < 16 * 256; i += 512) hsh[0][i] = (__bf16)0.0f;

  // async-prefetch x tile for t=0 into xsh[0] (wave 0; 2 x 32 lanes x 16B = 1KB)
  const int arow = lane >> 2;         // 0..7
  const int ach  = lane & 3;          // 16B chunk within 64B row
  if (w == 0) {
    for (int half = 0; half < 2; ++half) {
      const int r = arow + half * 8;
      const void* g = xin + ((size_t)(m0 + r) * 1024 + 0) * 32 + ach * 8;
      uint32_t l = (uint32_t)(uintptr_t)&xsh[0][r * 32 + ach * 8];
      async_g2l_b128(l, g);
    }
    wait_asynccnt0();
  }
  __syncthreads();

  const int c = w * 16 + mlo;         // column within H (and lane's B row)
  const float br_  = bih[c] + bhh[c];
  const float bz_  = bih[256 + c] + bhh[256 + c];
  const float bin_ = bih[512 + c];
  const float bhn_ = bhh[512 + c];

  // Pre-load r/z-gate Whh B fragments: stay in VGPRs for the entire loop.
  Frag bRz[16];
#pragma unroll
  for (int kc = 0; kc < 8; ++kc) {
    bRz[2 * kc + 0] = load_b(whhb + (size_t)(c)       * 256 + kc * 32 + bbase);
    bRz[2 * kc + 1] = load_b(whhb + (size_t)(256 + c) * 256 + kc * 32 + bbase);
  }

  int cur = 0;
  for (int t = 0; t < 1024; ++t) {
    const int nxt = cur ^ 1;

    // kick off async copy of next x tile into xsh[nxt] (wave 0 only)
    if (w == 0) {
      for (int half = 0; half < 2; ++half) {
        const int r = arow + half * 8;
        const void* g = xin + ((size_t)(m0 + r) * 1024 + (t + 1)) * 32 + ach * 8;
        uint32_t l = (uint32_t)(uintptr_t)&xsh[nxt][r * 32 + ach * 8];
        async_g2l_b128(l, g);
      }
      __builtin_prefetch(xin + ((size_t)(m0 + arow) * 1024 + (t + 2)) * 32, 0, 1);
    }

    const int zo = opaque_zero();     // loop-variant zero (defeats LICM only)

    // --- x fragment (A: 16x32, K = Ce = 32 exactly) from LDS staging ---
    Frag ax = load_a(&xsh[cur][mlo * 32 + base0]);

    v8f accR  = {0,0,0,0,0,0,0,0};
    v8f accZ  = {0,0,0,0,0,0,0,0};
    v8f accIN = {0,0,0,0,0,0,0,0};
    v8f accHN = {0,0,0,0,0,0,0,0};

    // gi = x @ Wih.T (one K-chunk); re-fetched each step (L0-resident)
    {
      Frag bR = load_b(wihb + (size_t)(c)       * 32 + bbase + zo);
      Frag bZ = load_b(wihb + (size_t)(256 + c) * 32 + bbase + zo);
      Frag bN = load_b(wihb + (size_t)(512 + c) * 32 + bbase + zo);
      accR  = wmma_bf16(ax, bR, accR);
      accZ  = wmma_bf16(ax, bZ, accZ);
      accIN = wmma_bf16(ax, bN, accIN);
    }
    // gh = h @ Whh.T (K = 256 -> 8 chunks); r/z B frags from registers,
    // n-gate B frags re-fetched. A frags double-buffered from LDS.
    Frag ah = load_a(&hsh[cur][mlo * 256 + base0]);
#pragma unroll
    for (int kc = 0; kc < 8; ++kc) {
      Frag ahn;
      if (kc < 7) ahn = load_a(&hsh[cur][mlo * 256 + (kc + 1) * 32 + base0]);
      Frag bN = load_b(whhb + (size_t)(512 + c) * 256 + kc * 32 + bbase + zo);
      accR  = wmma_bf16(ah, bRz[2 * kc + 0], accR);
      accZ  = wmma_bf16(ah, bRz[2 * kc + 1], accZ);
      accHN = wmma_bf16(ah, bN, accHN);
      if (kc < 7) ah = ahn;
    }
    // --- gate math (C layout: VGPR i -> row i + 8*hi, lane&15 -> col) ---
#pragma unroll
    for (int i = 0; i < 8; ++i) {
      const int m = i + hi * 8;
      float r  = sigm(accR[i] + br_);
      float z  = sigm(accZ[i] + bz_);
      float nn = tanhf(accIN[i] + bin_ + r * (accHN[i] + bhn_));
      float hold = (float)hsh[cur][m * 256 + c];
      float hnew = (1.0f - z) * nn + z * hold;
      hsh[nxt][m * 256 + c] = (__bf16)hnew;
      if (t == 1023) hout[(size_t)(m0 + m) * 256 + c] = hnew;
    }
    if (w == 0) wait_asynccnt0();   // x tile for t+1 resident before barrier
    __syncthreads();
    cur = nxt;
  }
}

// enc_out[b] = h_enc[b,:] . Wo + bo
__global__ void enc_head(const float* __restrict__ h, const float* __restrict__ wo,
                         const float* __restrict__ bo, float* __restrict__ eo) {
  int b = threadIdx.x;
  const float* hr = h + (size_t)b * 256;
  float s = bo[0];
  for (int k = 0; k < 256; ++k) s += hr[k] * wo[k];
  eo[b] = s;
}

// ---------------------------------------------------------------------------
// Decoder: single 512-thread block (BN couples whole batch each step).
// Wave w owns batch m-tile w; loops over 16 n-tiles. h ping-pongs in global ws.
// ---------------------------------------------------------------------------
__global__ __launch_bounds__(512) void dec_recurrence(
    const __bf16* __restrict__ wihb,   // [768][64] bf16 (K padded 33->64)
    const __bf16* __restrict__ whhb,   // [768][256] bf16
    const float* __restrict__ bih, const float* __restrict__ bhh,
    const float* __restrict__ wo, const float* __restrict__ bo,
    const float* __restrict__ decwg,   // [256][48][32] f32
    const float* __restrict__ encout,  // [256]
    const float* __restrict__ bn1g, const float* __restrict__ bn1b,
    const float* __restrict__ bnTg, const float* __restrict__ bnTb,  // [48][33]
    float* __restrict__ hbuf,          // [2][256][256] f32, [0] preloaded with h_enc
    float* __restrict__ out)           // [256][48]
{
  __shared__ __attribute__((aligned(16))) __bf16 dinp[256 * 64];
  __shared__ float ocol[256];
  const int tid  = threadIdx.x;
  const int lane = tid & 31;
  const int w    = tid >> 5;          // m-tile 0..15
  const int mlo  = lane & 15;
  const int hi   = lane >> 4;
  const int base0 = hi ? 8 : 0;
  const int bbase = hi ? 8 : 0;

  for (int i = tid; i < 256 * 64; i += 512) dinp[i] = (__bf16)0.0f;
  __syncthreads();

  // initial input: BN_1 over concat(dec_wg[:,0,:], enc_out)
  if (tid < 33) {
    const int cc = tid;
    float mu = 0.f;
    for (int b = 0; b < 256; ++b)
      mu += (cc < 32) ? decwg[(size_t)(b * 48) * 32 + cc] : encout[b];
    mu *= (1.0f / 256.0f);
    float var = 0.f;
    for (int b = 0; b < 256; ++b) {
      float x = (cc < 32) ? decwg[(size_t)(b * 48) * 32 + cc] : encout[b];
      float d = x - mu;
      var += d * d;
    }
    var *= (1.0f / 256.0f);
    float inv = rsqrtf(var + 1e-5f);
    float g = bn1g[cc], bb = bn1b[cc];
    for (int b = 0; b < 256; ++b) {
      float x = (cc < 32) ? decwg[(size_t)(b * 48) * 32 + cc] : encout[b];
      dinp[b * 64 + cc] = (__bf16)(g * (x - mu) * inv + bb);
    }
  }
  __syncthreads();

  int cur = 0;
  for (int t = 0; t < 48; ++t) {
    const int nxt = cur ^ 1;
    const float* hc = hbuf + (size_t)cur * 256 * 256;
    float*       hn = hbuf + (size_t)nxt * 256 * 256;

    for (int jj = 0; jj < 16; ++jj) {
      const int ccol = jj * 16 + mlo;
      v8f accR  = {0,0,0,0,0,0,0,0};
      v8f accZ  = {0,0,0,0,0,0,0,0};
      v8f accIN = {0,0,0,0,0,0,0,0};
      v8f accHN = {0,0,0,0,0,0,0,0};
      // gi over padded K=64
#pragma unroll
      for (int kc = 0; kc < 2; ++kc) {
        Frag ad = load_a(&dinp[(w * 16 + mlo) * 64 + kc * 32 + base0]);
        Frag bR = load_b(wihb + (size_t)(ccol)       * 64 + kc * 32 + bbase);
        Frag bZ = load_b(wihb + (size_t)(256 + ccol) * 64 + kc * 32 + bbase);
        Frag bN = load_b(wihb + (size_t)(512 + ccol) * 64 + kc * 32 + bbase);
        accR  = wmma_bf16(ad, bR, accR);
        accZ  = wmma_bf16(ad, bZ, accZ);
        accIN = wmma_bf16(ad, bN, accIN);
      }
      // gh over K=256 (h kept f32 in global, converted on load)
#pragma unroll 2
      for (int kc = 0; kc < 8; ++kc) {
        Frag ah;
        const float* hp = hc + (size_t)(w * 16 + mlo) * 256 + kc * 32 + base0;
#pragma unroll
        for (int e = 0; e < 8; ++e) {
          ah.e[e]     = (__bf16)hp[e];
          ah.e[8 + e] = (__bf16)hp[16 + e];
        }
        Frag bR = load_b(whhb + (size_t)(ccol)       * 256 + kc * 32 + bbase);
        Frag bZ = load_b(whhb + (size_t)(256 + ccol) * 256 + kc * 32 + bbase);
        Frag bN = load_b(whhb + (size_t)(512 + ccol) * 256 + kc * 32 + bbase);
        accR  = wmma_bf16(ah, bR, accR);
        accZ  = wmma_bf16(ah, bZ, accZ);
        accHN = wmma_bf16(ah, bN, accHN);
      }
      const float br_  = bih[ccol] + bhh[ccol];
      const float bz_  = bih[256 + ccol] + bhh[256 + ccol];
      const float bin_ = bih[512 + ccol];
      const float bhn_ = bhh[512 + ccol];
#pragma unroll
      for (int i = 0; i < 8; ++i) {
        const int row = w * 16 + i + hi * 8;
        float r  = sigm(accR[i] + br_);
        float z  = sigm(accZ[i] + bz_);
        float nn = tanhf(accIN[i] + bin_ + r * (accHN[i] + bhn_));
        float hold = hc[(size_t)row * 256 + ccol];
        hn[(size_t)row * 256 + ccol] = (1.0f - z) * nn + z * hold;
      }
    }
    __threadfence();
    __syncthreads();

    // output head
    if (tid < 256) {
      const float* hr = hn + (size_t)tid * 256;
      float s = bo[0];
      for (int k = 0; k < 256; ++k) s += hr[k] * wo[k];
      out[tid * 48 + t] = s;
      ocol[tid] = s;
    }
    __syncthreads();

    // BN for next step input
    if (t < 47 && tid < 33) {
      const int cc = tid;
      float mu = 0.f;
      for (int b = 0; b < 256; ++b)
        mu += (cc < 32) ? decwg[(size_t)(b * 48 + t + 1) * 32 + cc] : ocol[b];
      mu *= (1.0f / 256.0f);
      float var = 0.f;
      for (int b = 0; b < 256; ++b) {
        float x = (cc < 32) ? decwg[(size_t)(b * 48 + t + 1) * 32 + cc] : ocol[b];
        float d = x - mu;
        var += d * d;
      }
      var *= (1.0f / 256.0f);
      float inv = rsqrtf(var + 1e-5f);
      float g = bnTg[t * 33 + cc], bb = bnTb[t * 33 + cc];
      for (int b = 0; b < 256; ++b) {
        float x = (cc < 32) ? decwg[(size_t)(b * 48 + t + 1) * 32 + cc] : ocol[b];
        dinp[b * 64 + cc] = (__bf16)(g * (x - mu) * inv + bb);
      }
    }
    __syncthreads();
    cur = nxt;
  }
}

// ---------------------------------------------------------------------------
extern "C" void kernel_launch(void* const* d_in, const int* in_sizes, int n_in,
                              void* d_out, int out_size, void* d_ws, size_t ws_size,
                              hipStream_t stream) {
  (void)in_sizes; (void)n_in; (void)out_size; (void)ws_size;
  const float* X_enc  = (const float*)d_in[0];
  const float* X_dec  = (const float*)d_in[1];
  const float* Xg_enc = (const float*)d_in[2];
  const float* Xg_dec = (const float*)d_in[3];
  const float* Wg_e   = (const float*)d_in[4];
  const float* bg_e   = (const float*)d_in[5];
  const float* Wg_d   = (const float*)d_in[6];
  const float* bg_d   = (const float*)d_in[7];
  const float* eWih   = (const float*)d_in[8];
  const float* eWhh   = (const float*)d_in[9];
  const float* ebih   = (const float*)d_in[10];
  const float* ebhh   = (const float*)d_in[11];
  const float* eWo    = (const float*)d_in[12];
  const float* ebo    = (const float*)d_in[13];
  const float* dWih   = (const float*)d_in[14];
  const float* dWhh   = (const float*)d_in[15];
  const float* dbih   = (const float*)d_in[16];
  const float* dbhh   = (const float*)d_in[17];
  const float* dWo    = (const float*)d_in[18];
  const float* dbo    = (const float*)d_in[19];
  const float* bn1g   = (const float*)d_in[20];
  const float* bn1b   = (const float*)d_in[21];
  const float* bnTg   = (const float*)d_in[22];
  const float* bnTb   = (const float*)d_in[23];

  char* p = (char*)d_ws;
  __bf16* enc_in = (__bf16*)p; p += (size_t)256 * 1024 * 32 * 2;  // 16 MB
  __bf16* eWihb  = (__bf16*)p; p += (size_t)768 * 32 * 2;
  __bf16* eWhhb  = (__bf16*)p; p += (size_t)768 * 256 * 2;
  __bf16* dWihb  = (__bf16*)p; p += (size_t)768 * 64 * 2;
  __bf16* dWhhb  = (__bf16*)p; p += (size_t)768 * 256 * 2;
  float*  decwg  = (float*)p;  p += (size_t)256 * 48 * 32 * 4;
  float*  hbuf   = (float*)p;  p += (size_t)2 * 256 * 256 * 4;    // [0] = h_enc
  float*  encout = (float*)p;  p += (size_t)256 * 4;

  prep_weights<<<768, 256, 0, stream>>>(eWih, eWhh, dWih, dWhh, eWihb, eWhhb, dWihb, dWhhb);
  prep_enc_in<<<1024, 256, 0, stream>>>(X_enc, Xg_enc, Wg_e, bg_e, enc_in);
  prep_dec_wg<<<48, 256, 0, stream>>>(X_dec, Xg_dec, Wg_d, bg_d, decwg);
  enc_recurrence<<<16, 512, 0, stream>>>(enc_in, eWihb, eWhhb, ebih, ebhh, hbuf);
  enc_head<<<1, 256, 0, stream>>>(hbuf, eWo, ebo, encout);
  dec_recurrence<<<1, 512, 0, stream>>>(dWihb, dWhhb, dbih, dbhh, dWo, dbo, decwg, encout,
                                        bn1g, bn1b, bnTg, bnTb, hbuf, (float*)d_out);
}
